// SingleLSTMEncoder_18863496364485
// MI455X (gfx1250) — matrix operand order repaired and checked by reading, not compiled, not run on hardware
//
#include <hip/hip_runtime.h>
#include <math.h>

typedef unsigned int u32;
typedef __attribute__((ext_vector_type(16))) _Float16 v16h;
typedef __attribute__((ext_vector_type(8)))  _Float16 v8h;
typedef __attribute__((ext_vector_type(8)))  float    v8f;
typedef __attribute__((ext_vector_type(4)))  int      v4i;

#define B_   256
#define T_   512
#define CIN_ 64
#define H_   256
#define NC_  6
#define G4_  1024

// ---------------- workspace layout (bytes) ----------------
#define OFF_XBF   ((size_t)0)
#define SZ_XBF    ((size_t)T_*B_*CIN_*2)              // 16 MiB  x in f16, [t][b][64]
#define OFF_H1    (OFF_XBF + SZ_XBF)
#define SZ_H1     ((size_t)T_*B_*H_*2)                // 64 MiB  layer0 h sequence, [t][b][256]
#define OFF_PING  (OFF_H1 + SZ_H1)
#define SZ_PING   ((size_t)2*B_*H_*2)                 // layer1 h ping-pong
#define OFF_PK0   (OFF_PING + SZ_PING)
#define SZ_PK0    ((size_t)(CIN_+H_)*G4_*2)           // packed [Wih0|Whh0] B-fragments
#define OFF_PK1   (OFF_PK0 + SZ_PK0)
#define SZ_PK1    ((size_t)(H_+H_)*G4_*2)             // packed [Wih1|Whh1]
#define OFF_B0    (OFF_PK1 + SZ_PK1)                  // bih0+bhh0 (f32, 1024)
#define OFF_B1    (OFF_B0 + (size_t)G4_*4)
#define OFF_C0    (OFF_B1 + (size_t)G4_*4)            // c state layer0 (f32, B*H)
#define OFF_C1    (OFF_C0 + (size_t)B_*H_*4)
#define OFF_BAR   (OFF_C1 + (size_t)B_*H_*4)          // barrier counter @ +0, generation @ +64

// ---- optional CDNA5 async global->LDS staging path ----
#if defined(__has_builtin)
#if __has_builtin(__builtin_amdgcn_global_load_async_to_lds_b128) && \
    __has_builtin(__builtin_amdgcn_s_wait_asynccnt)
#define HAVE_ASYNC_LDS 1
#endif
#endif
#ifndef HAVE_ASYNC_LDS
#define HAVE_ASYNC_LDS 0
#endif

#if HAVE_ASYNC_LDS
typedef __attribute__((address_space(3))) v4i v4i_lds;
__device__ __forceinline__ void async_copy_b128(const void* gsrc, void* ldst) {
  // arg0: generic v4i* (per clang diagnostic); arg1: LDS pointer — AS3 cast is an
  // exact match if the builtin wants AS3, and implicitly widens to flat otherwise.
  // Low 32 bits of a generic LDS address == LDS offset on amdgcn.
  __builtin_amdgcn_global_load_async_to_lds_b128(
      (v4i*)(size_t)gsrc, (v4i_lds*)(u32)(size_t)ldst, 0, 0);
}
#endif

// ---------------- helpers ----------------
__device__ __forceinline__ v8f wmma_f16(v16h a, v16h b, v8f c) {
  // D = A(16x32 f16) * B(32x16 f16) + C(16x16 f32)
  return __builtin_amdgcn_wmma_f32_16x16x32_f16(false, a, false, b, (short)0, c, false, false);
}

// 16-bit A-matrix 16x32 fragment: lane L holds row m=L&15;
// lanes 0-15 cover K = {0..7, 16..23}, lanes 16-31 cover K = {8..15, 24..31}
__device__ __forceinline__ v16h load_a32(const _Float16* row, int half) {
  v8h a0 = *(const v8h*)(row + half * 8);
  v8h a1 = *(const v8h*)(row + 16 + half * 8);
  return __builtin_shufflevector(a0, a1, 0,1,2,3,4,5,6,7,8,9,10,11,12,13,14,15);
}

__device__ __forceinline__ void grid_barrier(u32* cnt, u32* gen, u32 nblk) {
  __syncthreads();
  if (threadIdx.x == 0) {
    u32 g = __hip_atomic_load(gen, __ATOMIC_RELAXED, __HIP_MEMORY_SCOPE_AGENT);
    u32 a = __hip_atomic_fetch_add(cnt, 1u, __ATOMIC_ACQ_REL, __HIP_MEMORY_SCOPE_AGENT);
    if (a == nblk - 1u) {
      __hip_atomic_store(cnt, 0u, __ATOMIC_RELAXED, __HIP_MEMORY_SCOPE_AGENT);
      __hip_atomic_fetch_add(gen, 1u, __ATOMIC_ACQ_REL, __HIP_MEMORY_SCOPE_AGENT);
    } else {
      while (__hip_atomic_load(gen, __ATOMIC_ACQUIRE, __HIP_MEMORY_SCOPE_AGENT) == g)
        __builtin_amdgcn_s_sleep(1);
    }
  }
  __syncthreads();
}

// Fast activations on the serial critical path: hardware v_exp/v_rcp.
__device__ __forceinline__ float fast_sigmoid(float x) {
  return __builtin_amdgcn_rcpf(1.0f + __expf(-x));
}
__device__ __forceinline__ float fast_tanh(float x) {
  // tanh(x) = 2*sigmoid(2x) - 1
  return __builtin_fmaf(2.0f, __builtin_amdgcn_rcpf(1.0f + __expf(-2.0f * x)), -1.0f);
}

// ---------------- prep kernels ----------------
__global__ void zero_kernel(u32* p, size_t n) {
  size_t i = (size_t)blockIdx.x * blockDim.x + threadIdx.x;
  size_t st = (size_t)gridDim.x * blockDim.x;
  for (; i < n; i += st) p[i] = 0u;
}

// x [B][T][CIN] f32  ->  xbf [t][b][CIN] f16
__global__ void prep_x_kernel(const float* __restrict__ x, _Float16* __restrict__ xbf) {
  size_t N = (size_t)T_ * B_ * CIN_;
  size_t i = (size_t)blockIdx.x * blockDim.x + threadIdx.x;
  size_t st = (size_t)gridDim.x * blockDim.x;
  for (; i < N; i += st) {
    int d = (int)(i % CIN_);
    size_t tb = i / CIN_;
    int b = (int)(tb % B_);
    int t = (int)(tb / B_);
    xbf[i] = (_Float16)x[((size_t)b * T_ + t) * CIN_ + d];
  }
}

// Pack combined weights [Wih | Whh] (gate-major rows, K = Kx+H) into WMMA
// B-fragment order: element index = ((ntile*KB + kb)*32 + lane)*16 + e
// with n = ntile*16 + (lane&15), k = kb*32 + (e>>3)*16 + (lane>>4)*8 + (e&7).
__global__ void pack_w_kernel(const float* __restrict__ Wih, const float* __restrict__ Whh,
                              int Kx, _Float16* __restrict__ pk) {
  int KB = (Kx + H_) >> 5;
  size_t N = (size_t)G4_ * (size_t)(Kx + H_);
  size_t i = (size_t)blockIdx.x * blockDim.x + threadIdx.x;
  size_t st = (size_t)gridDim.x * blockDim.x;
  for (; i < N; i += st) {
    int e = (int)(i & 15);
    int lane = (int)((i >> 4) & 31);
    size_t r = i >> 9;
    int kb = (int)(r % KB);
    int nt = (int)(r / KB);
    int n = nt * 16 + (lane & 15);
    int k = kb * 32 + (e >> 3) * 16 + (lane >> 4) * 8 + (e & 7);
    float w = (k < Kx) ? Wih[(size_t)n * Kx + k] : Whh[(size_t)n * H_ + (k - Kx)];
    pk[i] = (_Float16)w;
  }
}

__global__ void bias_kernel(const float* __restrict__ bih, const float* __restrict__ bhh,
                            float* __restrict__ bsum) {
  int i = blockIdx.x * blockDim.x + threadIdx.x;
  if (i < G4_) bsum[i] = bih[i] + bhh[i];
}

// ---------------- persistent fused LSTM layer ----------------
// grid = 32 blocks x 256 threads (8 wave32s) = 256 waves.
// wave w = blk*8+wv: batch tile m0 = (w&15)*16, hidden-col strip ns = w>>4 = blk>>1.
// All 8 waves of a block share ns -> the block's 4*KB weight B-fragments
// (KB*4 KiB total: 40 KiB / 64 KiB) are staged once into LDS and reused for
// all 512 timesteps. Each wave accumulates the i,f,g,o 16x16 gate tiles for
// its (m0,ns) chunk, then applies the LSTM nonlinearity in-registers.
template <int KX, int LAYER0>
__global__ __launch_bounds__(256) void lstm_kernel(
    const _Float16* __restrict__ xin,
    const _Float16* __restrict__ hsrc, _Float16* __restrict__ hout,
    const _Float16* __restrict__ pk, const float* __restrict__ bias,
    float* __restrict__ cbuf,
    u32* bar_cnt, u32* bar_gen, u32 nblk)
{
  constexpr int KB  = (KX + H_) >> 5;   // 32-wide K blocks total
  constexpr int KBX = KX >> 5;          // K blocks from the input segment

  extern __shared__ _Float16 lds_pk[];  // [4][KB][32][16]

  const int lane = threadIdx.x & 31;
  const int wave = threadIdx.x >> 5;
  const int w    = blockIdx.x * 8 + wave;     // 0..255
  const int m0   = (w & 15) * 16;
  const int ns   = w >> 4;                    // 0..15 (uniform per block)
  const int half = lane >> 4;
  const int nl   = lane & 15;
  const int mrow = m0 + nl;
  const int n    = ns * 16 + nl;

  // ---- stage this block's weight fragments into LDS (once) ----
  {
    constexpr int NPK = 4 * KB * 512;         // halves
    const int bns = blockIdx.x >> 1;          // ns for this block
    for (int c = threadIdx.x * 8; c < NPK; c += 256 * 8) {
      int q   = c / (KB * 512);
      int rem = c - q * (KB * 512);
      const _Float16* src = pk + ((size_t)(q * 16 + bns) * KB) * 512 + rem;
      _Float16*       dst = lds_pk + q * (KB * 512) + rem;
#if HAVE_ASYNC_LDS
      async_copy_b128(src, dst);
#else
      *(v8h*)dst = *(const v8h*)src;
#endif
    }
#if HAVE_ASYNC_LDS
    __builtin_amdgcn_s_wait_asynccnt(0);
#endif
    __syncthreads();
  }

  const float bi = bias[0 * H_ + n];
  const float bf = bias[1 * H_ + n];
  const float bg = bias[2 * H_ + n];
  const float bo = bias[3 * H_ + n];

  const int laneoff = lane * 16;

  for (int t = 0; t < T_; ++t) {
    v8f ai = {}, af = {}, ag = {}, ao = {};

    // input contribution: K = [0, KX)
    {
      const _Float16* row = xin + ((size_t)t * B_ + mrow) * (size_t)KX;
      #pragma unroll
      for (int kb = 0; kb < KBX; ++kb) {
        v16h a = load_a32(row + kb * 32, half);
        const _Float16* lb = lds_pk + kb * 512 + laneoff;
        ai = wmma_f16(a, *(const v16h*)(lb + 0 * KB * 512), ai);
        af = wmma_f16(a, *(const v16h*)(lb + 1 * KB * 512), af);
        ag = wmma_f16(a, *(const v16h*)(lb + 2 * KB * 512), ag);
        ao = wmma_f16(a, *(const v16h*)(lb + 3 * KB * 512), ao);
      }
    }
    // recurrent contribution: K = [KX, KX+H) using h_{t-1} (zero at t==0 -> skip)
    if (t > 0) {
      const _Float16* hrow;
      if (LAYER0) hrow = hsrc + ((size_t)(t - 1) * B_ + mrow) * (size_t)H_;
      else        hrow = hsrc + ((size_t)((t - 1) & 1) * B_ + mrow) * (size_t)H_;
      #pragma unroll
      for (int kb = KBX; kb < KB; ++kb) {
        v16h a = load_a32(hrow + (kb - KBX) * 32, half);
        const _Float16* lb = lds_pk + kb * 512 + laneoff;
        ai = wmma_f16(a, *(const v16h*)(lb + 0 * KB * 512), ai);
        af = wmma_f16(a, *(const v16h*)(lb + 1 * KB * 512), af);
        ag = wmma_f16(a, *(const v16h*)(lb + 2 * KB * 512), ag);
        ao = wmma_f16(a, *(const v16h*)(lb + 3 * KB * 512), ao);
      }
    }

    // fused elementwise LSTM cell update (D-layout: elem v -> m = m0 + half*8 + v, col n)
    _Float16* ho = hout + (LAYER0 ? (size_t)t * B_ * H_ : (size_t)(t & 1) * B_ * H_);
    #pragma unroll
    for (int v = 0; v < 8; ++v) {
      int m = m0 + half * 8 + v;
      float si = fast_sigmoid(ai[v] + bi);
      float sf = fast_sigmoid(af[v] + bf);
      float tg = fast_tanh(ag[v] + bg);
      float so = fast_sigmoid(ao[v] + bo);
      size_t ci = (size_t)m * H_ + n;
      float cn = sf * cbuf[ci] + si * tg;
      cbuf[ci] = cn;
      ho[ci] = (_Float16)(so * fast_tanh(cn));
    }

    grid_barrier(bar_cnt, bar_gen, nblk);
  }
}

// ---------------- FC + softmax on last step of layer1 ----------------
__global__ void fc_softmax_kernel(const _Float16* __restrict__ h2,
                                  const float* __restrict__ Wfc,
                                  const float* __restrict__ bfc,
                                  float* __restrict__ out) {
  int b = threadIdx.x;
  if (b >= B_) return;
  float lg[NC_];
  float mx = -3.4e38f;
  for (int c = 0; c < NC_; ++c) {
    float s = bfc[c];
    const float* wr = Wfc + (size_t)c * H_;
    for (int k = 0; k < H_; ++k) s += (float)h2[(size_t)b * H_ + k] * wr[k];
    lg[c] = s;
    mx = fmaxf(mx, s);
  }
  float den = 0.0f;
  for (int c = 0; c < NC_; ++c) { float e = __expf(lg[c] - mx); lg[c] = e; den += e; }
  float r = 1.0f / den;
  for (int c = 0; c < NC_; ++c) out[(size_t)b * NC_ + c] = lg[c] * r;
}

// ---------------- host side ----------------
extern "C" void kernel_launch(void* const* d_in, const int* in_sizes, int n_in,
                              void* d_out, int out_size, void* d_ws, size_t ws_size,
                              hipStream_t stream) {
  (void)in_sizes; (void)n_in; (void)out_size; (void)ws_size;

  const float* x    = (const float*)d_in[0];
  const float* Wih0 = (const float*)d_in[1];
  const float* Whh0 = (const float*)d_in[2];
  const float* bih0 = (const float*)d_in[3];
  const float* bhh0 = (const float*)d_in[4];
  const float* Wih1 = (const float*)d_in[5];
  const float* Whh1 = (const float*)d_in[6];
  const float* bih1 = (const float*)d_in[7];
  const float* bhh1 = (const float*)d_in[8];
  const float* Wfc  = (const float*)d_in[9];
  const float* bfc  = (const float*)d_in[10];

  char* ws = (char*)d_ws;
  _Float16* xbf  = (_Float16*)(ws + OFF_XBF);
  _Float16* h1   = (_Float16*)(ws + OFF_H1);
  _Float16* ping = (_Float16*)(ws + OFF_PING);
  _Float16* pk0  = (_Float16*)(ws + OFF_PK0);
  _Float16* pk1  = (_Float16*)(ws + OFF_PK1);
  float*    b0   = (float*)(ws + OFF_B0);
  float*    b1   = (float*)(ws + OFF_B1);
  float*    c0   = (float*)(ws + OFF_C0);
  float*    c1   = (float*)(ws + OFF_C1);
  u32*      bar  = (u32*)(ws + OFF_BAR);

  // zero c-states + barrier words (ws is poisoned by the harness)
  zero_kernel<<<256, 256, 0, stream>>>((u32*)(ws + OFF_C0),
                                       ((size_t)2 * B_ * H_ * 4 + 128) / 4);
  prep_x_kernel<<<2048, 256, 0, stream>>>(x, xbf);
  pack_w_kernel<<<1024, 256, 0, stream>>>(Wih0, Whh0, CIN_, pk0);
  pack_w_kernel<<<1024, 256, 0, stream>>>(Wih1, Whh1, H_,   pk1);
  bias_kernel<<<4, 256, 0, stream>>>(bih0, bhh0, b0);
  bias_kernel<<<4, 256, 0, stream>>>(bih1, bhh1, b1);

  // layer 0: input = xbf (K=64), recurrent h stored per-t in h1 (layer1's input)
  // LDS = 4 gates * KB * 1 KiB
  lstm_kernel<CIN_, 1><<<32, 256, ((CIN_ + H_) >> 5) * 4096, stream>>>(
      xbf, h1, h1, pk0, b0, c0, bar, bar + 16, 32u);
  // layer 1: input = h1 (K=256), recurrent h ping-pongs (only last step needed)
  lstm_kernel<H_, 0><<<32, 256, ((H_ + H_) >> 5) * 4096, stream>>>(
      h1, ping, ping, pk1, b1, c1, bar, bar + 16, 32u);

  fc_softmax_kernel<<<1, 256, 0, stream>>>(ping + (size_t)((T_ - 1) & 1) * B_ * H_,
                                           Wfc, bfc, (float*)d_out);
}
